// AttentiveRNNLanguageModel_50869592655367
// MI455X (gfx1250) — compile-verified, matrix operand order checked by reading, not run on hardware
//
#include <hip/hip_runtime.h>
#include <hip/hip_bf16.h>

// ---------------------------------------------------------------------------
// Types for CDNA5 WMMA (wave32): v_wmma_f32_16x16x32_bf16
// ---------------------------------------------------------------------------
typedef __attribute__((ext_vector_type(16))) __bf16 v16bf;
typedef __attribute__((ext_vector_type(8)))  __bf16 v8bf;
typedef __attribute__((ext_vector_type(8)))  float  v8f;

#if __has_builtin(__builtin_amdgcn_tensor_load_to_lds)
#define USE_TDM 1
typedef __attribute__((ext_vector_type(4))) unsigned u32x4;
typedef __attribute__((ext_vector_type(8))) int      i32x8;
typedef __attribute__((ext_vector_type(4))) int      i32x4;
#else
#define USE_TDM 0
#endif

// Problem constants
#define BB 4
#define TT 2048
#define HH 512
#define VV 32000
#define MM (BB * TT)     // 8192
#define FH (4 * HH)      // 2048

__device__ __forceinline__ __bf16 f2bf(float f) {
    union { float f; unsigned u; } v; v.f = f;
    unsigned r = v.u + 0x7FFFu + ((v.u >> 16) & 1u);   // round-to-nearest-even
    unsigned short s = (unsigned short)(r >> 16);
    __bf16 b; __builtin_memcpy(&b, &s, 2); return b;
}

__device__ __forceinline__ float sigm(float x) { return 1.f / (1.f + __expf(-x)); }

// Build one 16-bit A/B WMMA fragment (16 bf16 per lane) from a K-contiguous
// row. ISA layout: lanes 0-15 hold K {0..7, 16..23}; lanes 16-31 hold
// K {8..15, 24..31} of the 32-wide K window starting at `rowptr`.
__device__ __forceinline__ v16bf frag16(const __bf16* rowptr, int lane) {
    const int klo = (lane & 16) ? 8 : 0;
    v8bf lo = *(const v8bf*)(rowptr + klo);
    v8bf hi = *(const v8bf*)(rowptr + klo + 16);
    return __builtin_shufflevector(lo, hi, 0,1,2,3,4,5,6,7,8,9,10,11,12,13,14,15);
}

// ---------------------------------------------------------------------------
// Generic bf16 WMMA GEMM:  C[M,N] = act( A[M,K] @ W[N,K]^T + bias[N] )
// Block: 256 threads (8 waves, 2x4), tile 64(M) x 128(N), K-step 32.
// B-tile (128x32 bf16) is DMA'd into LDS by the Tensor Data Mover with the
// descriptor's pad fields reproducing the 112-byte padded row stride
// (16 dwords data + 12 dwords pad); tracked with TENSORcnt.
// ACT: 0 = none, 1 = tanh.  OUT_BF: 0 -> f32 C, 1 -> bf16 C.
// ---------------------------------------------------------------------------
#define GBM 64
#define GBN 128
#define GBK 32
#define GBKP 56   // padded row: 112 bytes -> 16B-aligned, bank-conflict-free

template<int ACT, int OUT_BF>
__global__ __launch_bounds__(256)
void gemm_bf16_kernel(const __bf16* __restrict__ A, const __bf16* __restrict__ W,
                      const float* __restrict__ bias,
                      float* __restrict__ Cf, __bf16* __restrict__ Cb,
                      int M, int N, int K)
{
    __shared__ __align__(16) __bf16 sA[GBM][GBKP];
    __shared__ __align__(16) __bf16 sB[GBN][GBKP];

    const int tid  = threadIdx.x;
    const int lane = tid & 31;
    const int wid  = tid >> 5;
    const int wm   = wid & 1;        // 0..1 -> 32-row half
    const int wn   = wid >> 1;       // 0..3 -> 32-col quarter
    const int m0   = blockIdx.y * GBM;
    const int n0   = blockIdx.x * GBN;

    v8f acc[2][2] = {};

    const int arow = tid >> 2, acol = (tid & 3) * 8;  // 64 rows x 4 chunks

    for (int k0 = 0; k0 < K; k0 += GBK) {
#if USE_TDM
        // --- B tile via Tensor Data Mover: one 2-D descriptor, wave 0 issues.
        if (wid == 0) {
            const unsigned long long ga =
                (unsigned long long)(const void*)&W[(size_t)n0 * K + k0];
            const unsigned ldsoff = (unsigned)(size_t)(const void*)&sB[0][0];
            // D# group0: count=1 | lds_addr | global_addr[56:0] | type=2
            u32x4 g0 = { 1u, ldsoff, (unsigned)ga,
                         (unsigned)(ga >> 32) | (2u << 30) };
            // D# group1: data_size=1(2B), pad_enable, pad_interval=3 (16 dw),
            // pad_amount=11 (12 dw); tensor_dim0=K, tensor_dim1=N,
            // tile_dim0=32, tile_dim1=128, tensor_dim0_stride=K
            i32x8 g1 = { (int)((1u << 16) | (1u << 20) | (3u << 22) | (11u << 25)),
                         (int)(((unsigned)K & 0xFFFFu) << 16),
                         (int)(((unsigned)K >> 16) | (((unsigned)N & 0xFFFFu) << 16)),
                         (int)(((unsigned)N >> 16) | (32u << 16)),
                         (int)GBN,
                         (int)K,
                         0, 0 };
            i32x4 gz4 = { 0, 0, 0, 0 };
            i32x8 gz8 = { 0, 0, 0, 0, 0, 0, 0, 0 };
            __builtin_amdgcn_tensor_load_to_lds(g0, g1, gz4, gz4, gz8, 0);
        }
        // --- A tile cooperative load (16B per thread), overlaps the DMA
        *(v8bf*)&sA[arow][acol] =
            *(const v8bf*)&A[(size_t)(m0 + arow) * K + k0 + acol];
        if (k0 + GBK < K)
            __builtin_prefetch(&A[(size_t)(m0 + arow) * K + k0 + GBK + acol], 0, 0);
        if (wid == 0) __builtin_amdgcn_s_wait_tensorcnt(0);
        __syncthreads();
#else
        *(v8bf*)&sA[arow][acol] =
            *(const v8bf*)&A[(size_t)(m0 + arow) * K + k0 + acol];
        {
            int c = tid;
            *(v8bf*)&sB[c >> 2][(c & 3) * 8] =
                *(const v8bf*)&W[(size_t)(n0 + (c >> 2)) * K + k0 + (c & 3) * 8];
            c = tid + 256;
            *(v8bf*)&sB[c >> 2][(c & 3) * 8] =
                *(const v8bf*)&W[(size_t)(n0 + (c >> 2)) * K + k0 + (c & 3) * 8];
        }
        if (k0 + GBK < K)
            __builtin_prefetch(&A[(size_t)(m0 + arow) * K + k0 + GBK + acol], 0, 0);
        __syncthreads();
#endif

        const int rl = lane & 15;
        v16bf a0 = frag16(&sA[wm * 32 + rl][0], lane);
        v16bf a1 = frag16(&sA[wm * 32 + 16 + rl][0], lane);
        v16bf b0 = frag16(&sB[wn * 32 + rl][0], lane);
        v16bf b1 = frag16(&sB[wn * 32 + 16 + rl][0], lane);

        acc[0][0] = __builtin_amdgcn_wmma_f32_16x16x32_bf16(false, a0, false, b0, (short)0, acc[0][0], false, false);
        acc[0][1] = __builtin_amdgcn_wmma_f32_16x16x32_bf16(false, a0, false, b1, (short)0, acc[0][1], false, false);
        acc[1][0] = __builtin_amdgcn_wmma_f32_16x16x32_bf16(false, a1, false, b0, (short)0, acc[1][0], false, false);
        acc[1][1] = __builtin_amdgcn_wmma_f32_16x16x32_bf16(false, a1, false, b1, (short)0, acc[1][1], false, false);
        __syncthreads();
    }

    // epilogue: lanes 0-15 -> N=lane, M=vgpr; lanes 16-31 -> N=lane-16, M=8+vgpr
    const int nl = lane & 15;
    const int mh = (lane >> 4) * 8;
#pragma unroll
    for (int i = 0; i < 2; ++i)
#pragma unroll
        for (int j = 0; j < 2; ++j) {
            const int n = n0 + wn * 32 + j * 16 + nl;
            const float bv = bias ? bias[n] : 0.f;
#pragma unroll
            for (int r = 0; r < 8; ++r) {
                const int m = m0 + wm * 32 + i * 16 + mh + r;
                float v = acc[i][j][r] + bv;
                if (ACT == 1) v = tanhf(v);
                if (OUT_BF) Cb[(size_t)m * N + n] = f2bf(v);
                else        Cf[(size_t)m * N + n] = v;
            }
        }
}

// ---------------------------------------------------------------------------
// Distributed LSTM scan: 16 workgroups (one per WGP), 256 threads each.
// WGP w owns hidden units [w*32, w*32+32) -> 128 gate columns whose W_hh
// slice (128 x 512 bf16 = 128 KB) lives in LDS for the entire scan.
// Per step: load full h (4 KB bf16, double-buffered in global/L2), 8 wave
// WMMA chains vs the LDS slice, local gate update, write own h slice, then
// one grid barrier (L2 atomic counter + s_sleep spin).
// ---------------------------------------------------------------------------
#define NWG 16
#define UPB 32
#define LW  520   // padded bf16 row stride: 1040 B (16B-aligned, banks 4r%64)

__global__ __launch_bounds__(256)
void lstm_scan_kernel(const float* __restrict__ xg,      // [B,T,4H] (biases folded)
                      const __bf16* __restrict__ Whh,    // [4H,H] bf16
                      const float* __restrict__ c0,      // [B*H]
                      __bf16* __restrict__ hbuf0,        // [B*H] bf16 (t even in)
                      __bf16* __restrict__ hbuf1,        // [B*H] bf16 (t odd  in)
                      float* __restrict__ enc,           // [B,T,H] f32
                      __bf16* __restrict__ encbf,        // [B,T,H] bf16
                      unsigned* __restrict__ cnt)        // grid-barrier counter
{
    __shared__ __align__(16) __bf16 sW[128][LW];   // resident W_hh slice
    __shared__ __align__(16) __bf16 shb[16][LW];   // h A-tile (rows 4-15 zero)
    __shared__ float sg[BB][128];
    __shared__ float sc[BB][UPB];

    const int tid  = threadIdx.x;
    const int lane = tid & 31;
    const int wv   = tid >> 5;          // wave 0..7 -> 16 gate columns each
    const int w    = blockIdx.x;        // 0..15
    const int col0 = w * UPB;

    // Load resident W_hh slice: local row r -> global row (r>>5)*H + col0 + (r&31)
    for (int c = tid; c < 128 * 64; c += 256) {
        const int r = c >> 6, cc = (c & 63) * 8;
        const int grow = (r >> 5) * HH + col0 + (r & 31);
        *(v8bf*)&sW[r][cc] = *(const v8bf*)&Whh[(size_t)grow * HH + cc];
    }
    for (int i = tid; i < 16 * LW; i += 256)
        ((unsigned short*)shb)[i] = 0;               // zero (rows 4-15 stay zero)
    if (tid < BB * UPB)
        sc[tid >> 5][tid & 31] = c0[(tid >> 5) * HH + col0 + (tid & 31)];
    __syncthreads();

    unsigned target = NWG;
    for (int t = 0; t < TT; ++t) {
        const __bf16* hin  = (t & 1) ? hbuf1 : hbuf0;
        __bf16*       hout = (t & 1) ? hbuf0 : hbuf1;

        // pull full h (4x512 bf16) into A-tile rows 0..3 (one 16B chunk/thread)
        {
            const int r = tid >> 6, cc = (tid & 63) * 8;
            *(v8bf*)&shb[r][cc] = *(const v8bf*)&hin[r * HH + cc];
        }
        __syncthreads();

        // opaque zero offset: defeat cross-step CSE/LICM of sW fragments
        unsigned woff = 0;
        asm volatile("" : "+v"(woff));

        v8f acc = {};
        const int rl = lane & 15;
        for (int k0 = 0; k0 < HH; k0 += 32) {
            v16bf a = frag16(&shb[rl][k0], lane);
            v16bf b = frag16(&sW[wv * 16 + rl][k0 + woff], lane);
            acc = __builtin_amdgcn_wmma_f32_16x16x32_bf16(false, a, false, b, (short)0, acc, false, false);
        }
        if (lane < 16) {
#pragma unroll
            for (int r = 0; r < BB; ++r) sg[r][wv * 16 + lane] = acc[r];  // rows 0..3 valid
        }
        __syncthreads();

        // local gate update for our 32 hidden units (gate blocks of 32 in sg)
        if (tid < BB * UPB) {
            const int b = tid >> 5, u = tid & 31;
            const int col = col0 + u;
            const float* xr = xg + ((size_t)b * TT + t) * FH;
            const float gi = xr[col]          + sg[b][u];
            const float gf = xr[HH + col]     + sg[b][32 + u];
            const float gg = xr[2 * HH + col] + sg[b][64 + u];
            const float go = xr[3 * HH + col] + sg[b][96 + u];
            const float cv = sigm(gf) * sc[b][u] + sigm(gi) * tanhf(gg);
            const float hn = sigm(go) * tanhf(cv);
            sc[b][u] = cv;
            const size_t eo = ((size_t)b * TT + t) * HH + col;
            const __bf16 hb = f2bf(hn);
            enc[eo]   = hn;
            encbf[eo] = hb;
            hout[b * HH + col] = hb;
        }
        __syncthreads();

        // grid barrier: release h slice, arrive, spin with s_sleep
        if (tid == 0) {
            __threadfence();
            __hip_atomic_fetch_add(cnt, 1u, __ATOMIC_ACQ_REL, __HIP_MEMORY_SCOPE_AGENT);
            while (__hip_atomic_load(cnt, __ATOMIC_ACQUIRE, __HIP_MEMORY_SCOPE_AGENT) < target)
                __builtin_amdgcn_s_sleep(1);
        }
        __syncthreads();
        target += NWG;
    }
}

__global__ void lstm_init_kernel(const float* __restrict__ h0,
                                 __bf16* __restrict__ hbuf0,
                                 unsigned* __restrict__ cnt) {
    int i = blockIdx.x * 256 + threadIdx.x;
    if (i < BB * HH) hbuf0[i] = f2bf(h0[i]);
    if (i == 0) *cnt = 0u;
}

// ---------------------------------------------------------------------------
// Small helper kernels
// ---------------------------------------------------------------------------
__global__ void cvt_bf16_kernel(const float* __restrict__ in, __bf16* __restrict__ out, int n) {
    int i = blockIdx.x * 256 + threadIdx.x;
    if (i < n) out[i] = f2bf(in[i]);
}

__global__ void bias_sum_kernel(const float* a, const float* b, float* o, int n) {
    int i = blockIdx.x * 256 + threadIdx.x;
    if (i < n) o[i] = a[i] + b[i];
}

__global__ void embed_kernel(const int* __restrict__ tok, const __bf16* __restrict__ embbf,
                             __bf16* __restrict__ xbf, int n) {
    int i = blockIdx.x * 256 + threadIdx.x;
    if (i >= n) return;
    int bt = i >> 9, h = i & (HH - 1);
    xbf[i] = embbf[(size_t)tok[bt] * HH + h];
}

// s[row] = dot(A1[row,:], wa2) + ba2 ; one wave32 per row
__global__ __launch_bounds__(256)
void score_kernel(const float* __restrict__ A1, const float* __restrict__ wa2,
                  const float* __restrict__ ba2, float* __restrict__ s, int rows) {
    const int wid = threadIdx.x >> 5, lane = threadIdx.x & 31;
    const int row = blockIdx.x * 8 + wid;
    if (row >= rows) return;
    const float* a = A1 + (size_t)row * HH;
    float sum = 0.f;
    for (int i = lane; i < HH; i += 32) sum += a[i] * wa2[i];
    for (int off = 16; off > 0; off >>= 1) sum += __shfl_xor(sum, off, 32);
    if (lane == 0) s[row] = sum + ba2[0];
}

__global__ __launch_bounds__(256)
void smax_kernel(const float* __restrict__ s, float* __restrict__ smax) {
    __shared__ float red[256];
    const int b = blockIdx.x;
    float m = -1e30f;
    for (int i = threadIdx.x; i < TT; i += 256) m = fmaxf(m, s[b * TT + i]);
    red[threadIdx.x] = m; __syncthreads();
    for (int w = 128; w > 0; w >>= 1) {
        if (threadIdx.x < w) red[threadIdx.x] = fmaxf(red[threadIdx.x], red[threadIdx.x + w]);
        __syncthreads();
    }
    if (threadIdx.x == 0) smax[b] = red[0];
}

// Causal softmax pooling via O(T) scan; thread (b,h) writes ctx as bf16 into
// the left half of the concatenated [ctx | enc] GEMM input.
__global__ __launch_bounds__(512)
void prefix_kernel(const float* __restrict__ s, const float* __restrict__ smax,
                   const float* __restrict__ enc, __bf16* __restrict__ catbf) {
    const int b = blockIdx.x, h = threadIdx.x;
    const float m = smax[b];
    float den = 0.f, num = 0.f;
    for (int t = 0; t < TT; ++t) {
        const float e = __expf(s[b * TT + t] - m);
        den += e;
        num += e * enc[(((size_t)b * TT + t) << 9) + h];
        catbf[(((size_t)b * TT + t) << 10) + h] = f2bf(num / den);
    }
}

__global__ void cat_enc_kernel(const float* __restrict__ enc, __bf16* __restrict__ catbf, int n) {
    int i = blockIdx.x * 256 + threadIdx.x;
    if (i >= n) return;
    int bt = i >> 9, h = i & (HH - 1);
    catbf[(((size_t)bt) << 10) + HH + h] = f2bf(enc[i]);
}

// ---------------------------------------------------------------------------
// Workspace layout (bytes, 256-aligned)
// ---------------------------------------------------------------------------
static constexpr size_t OFF_EMB   = 0;                        // V*H bf16  = 32,768,000
static constexpr size_t OFF_WIH   = OFF_EMB   + 32768000;     // 4H*H bf16 =  2,097,152
static constexpr size_t OFF_WHH   = OFF_WIH   + 2097152;
static constexpr size_t OFF_WA1   = OFF_WHH   + 2097152;      // H*H bf16  =    524,288
static constexpr size_t OFF_WC    = OFF_WA1   + 524288;       // H*2H bf16 =  1,048,576
static constexpr size_t OFF_XBF   = OFF_WC    + 1048576;      // M*H bf16  =  8,388,608
static constexpr size_t OFF_BSUM  = OFF_XBF   + 8388608;      // 4H f32    =      8,192
static constexpr size_t OFF_XG    = OFF_BSUM  + 8192;         // M*4H f32  = 67,108,864
static constexpr size_t OFF_ENC   = OFF_XG    + 67108864;     // M*H f32   = 16,777,216
static constexpr size_t OFF_ENCBF = OFF_ENC   + 16777216;     // M*H bf16  =  8,388,608
static constexpr size_t OFF_A1    = OFF_ENCBF + 8388608;      // M*H f32   = 16,777,216
static constexpr size_t OFF_S     = OFF_A1    + 16777216;     // M f32     =     32,768
static constexpr size_t OFF_SMAX  = OFF_S     + 32768;        //                    256
static constexpr size_t OFF_CAT   = OFF_SMAX  + 256;          // M*2H bf16 = 16,777,216
static constexpr size_t OFF_COMB  = OFF_CAT   + 16777216;     // M*H bf16  =  8,388,608
static constexpr size_t OFF_HB0   = OFF_COMB  + 8388608;      // B*H bf16  =      4,096
static constexpr size_t OFF_HB1   = OFF_HB0   + 4096;
static constexpr size_t OFF_CNT   = OFF_HB1   + 4096;         // counter

extern "C" void kernel_launch(void* const* d_in, const int* in_sizes, int n_in,
                              void* d_out, int out_size, void* d_ws, size_t ws_size,
                              hipStream_t stream) {
    (void)in_sizes; (void)n_in; (void)out_size; (void)ws_size;

    const int*   tok  = (const int*)  d_in[0];
    const float* h0   = (const float*)d_in[1];
    const float* c0   = (const float*)d_in[2];
    const float* emb  = (const float*)d_in[3];
    const float* Wih  = (const float*)d_in[4];
    const float* Whh  = (const float*)d_in[5];
    const float* bih  = (const float*)d_in[6];
    const float* bhh  = (const float*)d_in[7];
    const float* Wa1  = (const float*)d_in[8];
    const float* ba1  = (const float*)d_in[9];
    const float* wa2  = (const float*)d_in[10];
    const float* ba2  = (const float*)d_in[11];
    const float* Wc   = (const float*)d_in[12];
    const float* bc   = (const float*)d_in[13];
    const float* bdec = (const float*)d_in[14];

    char* ws = (char*)d_ws;
    __bf16* emb_bf  = (__bf16*)(ws + OFF_EMB);
    __bf16* wih_bf  = (__bf16*)(ws + OFF_WIH);
    __bf16* whh_bf  = (__bf16*)(ws + OFF_WHH);
    __bf16* wa1_bf  = (__bf16*)(ws + OFF_WA1);
    __bf16* wc_bf   = (__bf16*)(ws + OFF_WC);
    __bf16* x_bf    = (__bf16*)(ws + OFF_XBF);
    float*  bsum    = (float*) (ws + OFF_BSUM);
    float*  xg      = (float*) (ws + OFF_XG);
    float*  enc     = (float*) (ws + OFF_ENC);
    __bf16* enc_bf  = (__bf16*)(ws + OFF_ENCBF);
    float*  A1      = (float*) (ws + OFF_A1);
    float*  sv      = (float*) (ws + OFF_S);
    float*  smaxp   = (float*) (ws + OFF_SMAX);
    __bf16* cat_bf  = (__bf16*)(ws + OFF_CAT);
    __bf16* comb_bf = (__bf16*)(ws + OFF_COMB);
    __bf16* hb0     = (__bf16*)(ws + OFF_HB0);
    __bf16* hb1     = (__bf16*)(ws + OFF_HB1);
    unsigned* cntp  = (unsigned*)(ws + OFF_CNT);
    float*  out     = (float*) d_out;

    // 1) one-time fp32 -> bf16 weight conversions
    cvt_bf16_kernel<<<64000, 256, 0, stream>>>(emb, emb_bf, VV * HH);
    cvt_bf16_kernel<<<4096, 256, 0, stream>>>(Wih, wih_bf, FH * HH);
    cvt_bf16_kernel<<<4096, 256, 0, stream>>>(Whh, whh_bf, FH * HH);
    cvt_bf16_kernel<<<1024, 256, 0, stream>>>(Wa1, wa1_bf, HH * HH);
    cvt_bf16_kernel<<<2048, 256, 0, stream>>>(Wc, wc_bf, HH * 2 * HH);
    bias_sum_kernel<<<8, 256, 0, stream>>>(bih, bhh, bsum, FH);

    // 2) embedding gather (bf16)
    embed_kernel<<<16384, 256, 0, stream>>>(tok, emb_bf, x_bf, MM * HH);

    // 3) xg = x @ W_ih^T + (b_ih + b_hh)    [8192 x 2048]
    gemm_bf16_kernel<0, 0><<<dim3(FH / GBN, MM / GBM), 256, 0, stream>>>(
        x_bf, wih_bf, bsum, xg, nullptr, MM, FH, HH);

    // 4) distributed serial LSTM scan (16 WGPs, LDS-resident W_hh slices)
    lstm_init_kernel<<<8, 256, 0, stream>>>(h0, hb0, cntp);
    lstm_scan_kernel<<<NWG, 256, 0, stream>>>(xg, whh_bf, c0, hb0, hb1, enc, enc_bf, cntp);

    // 5) A1 = tanh(enc @ Wa1^T + ba1)       [8192 x 512]
    gemm_bf16_kernel<1, 0><<<dim3(HH / GBN, MM / GBM), 256, 0, stream>>>(
        enc_bf, wa1_bf, ba1, A1, nullptr, MM, HH, HH);

    // 6) scalar attention scores + global max + causal prefix pooling
    score_kernel<<<MM / 8, 256, 0, stream>>>(A1, wa2, ba2, sv, MM);
    smax_kernel<<<BB, 256, 0, stream>>>(sv, smaxp);
    prefix_kernel<<<BB, HH, 0, stream>>>(sv, smaxp, enc, cat_bf);
    cat_enc_kernel<<<16384, 256, 0, stream>>>(enc, cat_bf, MM * HH);

    // 7) combined = tanh([ctx|enc] @ Wc^T + bc) -> bf16   [8192 x 512], K=1024
    gemm_bf16_kernel<1, 1><<<dim3(HH / GBN, MM / GBM), 256, 0, stream>>>(
        cat_bf, wc_bf, bc, nullptr, comb_bf, MM, HH, 2 * HH);

    // 8) decoded = combined @ emb^T + b_dec  [8192 x 32000] -> d_out (f32)
    gemm_bf16_kernel<0, 0><<<dim3(VV / GBN, MM / GBM), 256, 0, stream>>>(
        comb_bf, emb_bf, bdec, out, nullptr, MM, VV, HH);
}
